// RWKVBlock_90580860273231
// MI455X (gfx1250) — compile-verified, matrix operand order checked by reading, not run on hardware
//
#include <hip/hip_runtime.h>

// ---------------- types ----------------
typedef __attribute__((ext_vector_type(16))) __bf16 v16bf;
typedef __attribute__((ext_vector_type(8)))  __bf16 v8bf;
typedef __attribute__((ext_vector_type(8)))  float  v8f;

#define CC 1024
#define TT 2048
#define BB 4
#define MM (BB * TT)   // 8192 rows

// CDNA5 direct global->LDS async copy path (guarded; falls back to VGPR bounce)
#if defined(__has_builtin)
#  if __has_builtin(__builtin_amdgcn_global_load_async_to_lds_b128) && \
      __has_builtin(__builtin_amdgcn_s_wait_asynccnt)
#    define USE_ASYNC_COPY 1
#  endif
#endif
#ifndef USE_ASYNC_COPY
#  define USE_ASYNC_COPY 0
#endif

#if USE_ASYNC_COPY
// builtin signature (per hipcc diagnostic): (as1 int4* src, as3 int4* dst, imm, imm)
typedef int v4i_gcc __attribute__((__vector_size__(16)));
typedef __attribute__((address_space(1))) v4i_gcc* as1_v4i;
typedef __attribute__((address_space(3))) v4i_gcc* as3_v4i;
#endif

// ---------------- helpers ----------------
__device__ __forceinline__ __bf16 f2bf(float f) {
    union { float f; unsigned u; } in; in.f = f;
    unsigned r = in.u + 0x7fffu + ((in.u >> 16) & 1u);   // round-to-nearest-even
    union { unsigned short s; __bf16 b; } out;
    out.s = (unsigned short)(r >> 16);
    return out.b;
}

__device__ __forceinline__ void stage_wait() {
#if USE_ASYNC_COPY
    __builtin_amdgcn_s_wait_asynccnt(0);
#endif
}

#if !USE_ASYNC_COPY
// fallback: copy 16 bf16 (32B) global -> LDS through VGPRs
__device__ __forceinline__ void stage_sync(const __bf16* __restrict__ g, __bf16* l) {
    v8bf c0 = *(const v8bf*)g;
    v8bf c1 = *(const v8bf*)(g + 8);
    *(v8bf*)l = c0;
    *(v8bf*)(l + 8) = c1;
}
#endif

// f32 -> bf16 elementwise convert (weights, once per launch)
__global__ void cvt_bf16_kernel(const float* __restrict__ in, __bf16* __restrict__ out, int n) {
    int i = blockIdx.x * blockDim.x + threadIdx.x;
    if (i < n) out[i] = f2bf(in[i]);
}

// ---------------- LayerNorm + time-shift mix ----------------
// mode 0: xxx = h + (h - shifted)*ts ; t==0 shifted = ts itself   (time mixing)
// mode 1: xxx = h + (shifted - h)*ts ; t==0 shifted = h[T-1]      (channel mixing)
__device__ void rowstats(const float* __restrict__ row, float* sh, float& mu, float& rstd) {
    int tid = threadIdx.x;
    float s = 0.f, s2 = 0.f;
    for (int i = tid; i < CC; i += 256) { float v = row[i]; s += v; s2 += v * v; }
    sh[tid] = s; sh[256 + tid] = s2;
    __syncthreads();
    for (int st = 128; st > 0; st >>= 1) {
        if (tid < st) { sh[tid] += sh[tid + st]; sh[256 + tid] += sh[256 + tid + st]; }
        __syncthreads();
    }
    mu = sh[0] * (1.0f / CC);
    float var = sh[256] * (1.0f / CC) - mu * mu;
    rstd = rsqrtf(var + 1e-5f);
    __syncthreads();
}

__global__ __launch_bounds__(256)
void ln_mix_kernel(const float* __restrict__ X, const float* __restrict__ w,
                   const float* __restrict__ bias, const float* __restrict__ ts,
                   __bf16* __restrict__ out, int mode) {
    __shared__ float sh[512];
    const int row = blockIdx.x;            // b*T + t
    const int t = row & (TT - 1);
    float mu, rstd, mup = 0.f, rstdp = 1.f;
    rowstats(X + (size_t)row * CC, sh, mu, rstd);
    bool havePrev = true; int prev = row - 1;
    if (t == 0) { if (mode == 0) havePrev = false; else prev = row + (TT - 1); }
    if (havePrev) rowstats(X + (size_t)prev * CC, sh, mup, rstdp);
    const float* xr = X + (size_t)row * CC;
    const float* xp = X + (size_t)(havePrev ? prev : row) * CC;
    for (int i = threadIdx.x; i < CC; i += 256) {
        float tsi = ts[i];
        float h   = (xr[i] - mu) * rstd * w[i] + bias[i];
        float shv = havePrev ? ((xp[i] - mup) * rstdp * w[i] + bias[i]) : tsi;
        float xxx = (mode == 0) ? (h + (h - shv) * tsi) : (h + (shv - h) * tsi);
        out[(size_t)row * CC + i] = f2bf(xxx);
    }
}

// ---------------- bf16 WMMA GEMM: out[M,N] = A[M,K] * W[N,K]^T ----------------
// Double-buffered LDS pipeline; async global->LDS staging of tile k+1 overlapped
// with WMMAs on tile k; one barrier per K-step; prefetch tile k+2.
// EPI 0: store f32 | 1: sigmoid f32 | 2: relu^2 -> bf16 | 3: out = aux1 + aux2*val (f32)
#define TM 128
#define TN 128
#define TK 32
#define ASTR 40   // LDS row stride in elements (80 bytes: 20 dwords, conflict-spread)

template <int EPI>
__global__ __launch_bounds__(256)
void gemm_bf16_kernel(const __bf16* __restrict__ A, const __bf16* __restrict__ Wt,
                      float* __restrict__ outF, __bf16* __restrict__ outB,
                      const float* __restrict__ aux1, const float* __restrict__ aux2,
                      int N, int K) {
    __shared__ __bf16 sA[2][TM * ASTR];   // 2 x 10 KB
    __shared__ __bf16 sB[2][TN * ASTR];   // 2 x 10 KB  (40 KB total)
    const int tid   = threadIdx.x;
    const int lane  = tid & 31;
    const int wave  = tid >> 5;           // 0..7
    const int m0    = blockIdx.y * TM;
    const int n0    = blockIdx.x * TN;
    const int wm    = (wave & 3) * 32;    // wave sub-tile: 32 rows
    const int wn    = (wave >> 2) * 64;   //                64 cols
    const int lrow  = lane & 15;
    const int lhalf = lane >> 4;

    v8f acc[2][4] = {};

    const int ldRow = tid >> 1;           // 0..127
    const int ldOff = (tid & 1) * 16;     // element offset 0/16
    const __bf16* gA = A  + (size_t)(m0 + ldRow) * K + ldOff;
    const __bf16* gB = Wt + (size_t)(n0 + ldRow) * K + ldOff;
    __bf16* la[2] = { &sA[0][ldRow * ASTR + ldOff], &sA[1][ldRow * ASTR + ldOff] };
    __bf16* lb[2] = { &sB[0][ldRow * ASTR + ldOff], &sB[1][ldRow * ASTR + ldOff] };

#if USE_ASYNC_COPY
    // hoist addrspace casts (and their null checks) out of the hot loop
    as1_v4i gA1 = (as1_v4i)(unsigned long long)gA;
    as1_v4i gB1 = (as1_v4i)(unsigned long long)gB;
    as3_v4i lA3[2] = { (as3_v4i)la[0], (as3_v4i)la[1] };
    as3_v4i lB3[2] = { (as3_v4i)lb[0], (as3_v4i)lb[1] };

    // prologue: stage K-tile 0 into buffer 0
    __builtin_amdgcn_global_load_async_to_lds_b128(gA1,     lA3[0],     0, 0);
    __builtin_amdgcn_global_load_async_to_lds_b128(gA1 + 1, lA3[0] + 1, 0, 0);
    __builtin_amdgcn_global_load_async_to_lds_b128(gB1,     lB3[0],     0, 0);
    __builtin_amdgcn_global_load_async_to_lds_b128(gB1 + 1, lB3[0] + 1, 0, 0);
#else
    stage_sync(gA, la[0]);
    stage_sync(gB, lb[0]);
#endif
    stage_wait();
    __syncthreads();

    int s = 0;
    for (int k0 = 0; k0 < K; k0 += TK) {
        // stage next K-tile into the other buffer while we compute on this one
        if (k0 + TK < K) {
#if USE_ASYNC_COPY
            const int kv = (k0 + TK) >> 3;           // v4i units (TK=32 -> exact)
            as3_v4i da = lA3[s ^ 1], db = lB3[s ^ 1];
            __builtin_amdgcn_global_load_async_to_lds_b128(gA1 + kv,     da,     0, 0);
            __builtin_amdgcn_global_load_async_to_lds_b128(gA1 + kv + 1, da + 1, 0, 0);
            __builtin_amdgcn_global_load_async_to_lds_b128(gB1 + kv,     db,     0, 0);
            __builtin_amdgcn_global_load_async_to_lds_b128(gB1 + kv + 1, db + 1, 0, 0);
#else
            stage_sync(gA + k0 + TK, la[s ^ 1]);
            stage_sync(gB + k0 + TK, lb[s ^ 1]);
#endif
        }
        // warm L2/WGP$ two tiles ahead (global_prefetch_b8)
        if (k0 + 2 * TK < K) {
            __builtin_prefetch(gA + k0 + 2 * TK, 0, 3);
            __builtin_prefetch(gB + k0 + 2 * TK, 0, 3);
        }

        // fragments per documented 16-bit layout:
        // lane holds K in {8*lhalf .. 8*lhalf+7} U {16+8*lhalf .. 16+8*lhalf+7}
        v16bf afrag[2], bfrag[4];
        for (int i = 0; i < 2; i++) {
            const __bf16* p = &sA[s][(wm + i * 16 + lrow) * ASTR] + lhalf * 8;
            ((v8bf*)&afrag[i])[0] = *(const v8bf*)p;
            ((v8bf*)&afrag[i])[1] = *(const v8bf*)(p + 16);
        }
        for (int j = 0; j < 4; j++) {
            const __bf16* p = &sB[s][(wn + j * 16 + lrow) * ASTR] + lhalf * 8;
            ((v8bf*)&bfrag[j])[0] = *(const v8bf*)p;
            ((v8bf*)&bfrag[j])[1] = *(const v8bf*)(p + 16);
        }
        for (int i = 0; i < 2; i++)
            for (int j = 0; j < 4; j++)
                acc[i][j] = __builtin_amdgcn_wmma_f32_16x16x32_bf16(
                    false, afrag[i], false, bfrag[j], (short)0, acc[i][j], false, false);

        stage_wait();       // this wave's async copies into buf s^1 have landed
        __syncthreads();    // everyone's have
        s ^= 1;
    }

    // epilogue; C/D layout: VGPR r -> m = r + 8*lhalf, n = lrow
    for (int i = 0; i < 2; i++) {
        for (int j = 0; j < 4; j++) {
            const int mb = m0 + wm + i * 16 + 8 * lhalf;
            const int nb = n0 + wn + j * 16 + lrow;
            for (int r = 0; r < 8; r++) {
                size_t idx = (size_t)(mb + r) * N + nb;
                float val = acc[i][j][r];
                if (EPI == 0) {
                    outF[idx] = val;
                } else if (EPI == 1) {
                    outF[idx] = 1.f / (1.f + __expf(-val));
                } else if (EPI == 2) {
                    float tpos = fmaxf(val, 0.f);
                    outB[idx] = f2bf(tpos * tpos);
                } else {
                    outF[idx] = aux1[idx] + aux2[idx] * val;
                }
            }
        }
    }
}

// ---------------- WKV log-sum-exp recurrence + residual ----------------
__global__ __launch_bounds__(256)
void wkv_kernel(const float* __restrict__ u, const float* __restrict__ Kb,
                const float* __restrict__ Vb, const float* __restrict__ Rb,
                const float* __restrict__ X, float* __restrict__ X1) {
    const int gid = blockIdx.x * blockDim.x + threadIdx.x;   // b*C + c
    const int c = gid & (CC - 1);
    const int b = gid >> 10;
    const float uu = u[c];
    float aa = 0.f, bb = 0.f;
    size_t base = (size_t)b * TT * CC + c;
    for (int t = 0; t < TT; t++) {
        size_t idx = base + (size_t)t * CC;
        float kk = Kb[idx], vv = Vb[idx];
        float p  = fmaxf(bb, uu + kk);
        float e1 = __expf(bb - p);
        float e2 = __expf(uu + kk - p);
        float out = (e1 * aa + e2 * vv) / (e1 + e2);
        float p2  = fmaxf(bb, kk);
        float e1b = __expf(bb - p2);
        float e2b = __expf(kk - p2);
        float na = e1b * aa + e2b * vv;
        float nb = p2 + __logf(e1b + e2b);
        aa = na; bb = nb;
        X1[idx] = X[idx] + Rb[idx] * out;
    }
}

// ---------------- launch ----------------
extern "C" void kernel_launch(void* const* d_in, const int* in_sizes, int n_in,
                              void* d_out, int out_size, void* d_ws, size_t ws_size,
                              hipStream_t stream) {
    (void)in_sizes; (void)n_in; (void)out_size; (void)ws_size;
    const float* x     = (const float*)d_in[0];
    // d_in[1] = time_decay (unused by reference math; mirrored)
    const float* tf    = (const float*)d_in[2];
    const float* ts    = (const float*)d_in[3];
    const float* Wk    = (const float*)d_in[4];
    const float* Wv    = (const float*)d_in[5];
    const float* Wr    = (const float*)d_in[6];
    const float* Wck   = (const float*)d_in[7];
    const float* Wcv   = (const float*)d_in[8];
    const float* Wcr   = (const float*)d_in[9];
    const float* ln1w  = (const float*)d_in[10];
    const float* ln1b  = (const float*)d_in[11];
    const float* ln2w  = (const float*)d_in[12];
    const float* ln2b  = (const float*)d_in[13];
    float* out = (float*)d_out;

    const size_t MBy = 1024ull * 1024ull;
    char* w = (char*)d_ws;
    __bf16* Abf  = (__bf16*)(w + 0);           // 16 MB: xxx / xxx2 (M x C bf16)
    float*  Kbuf = (float*)(w + 16 * MBy);     // 32 MB
    float*  Vbuf = (float*)(w + 48 * MBy);     // 32 MB
    float*  Rbuf = (float*)(w + 80 * MBy);     // 32 MB (sigmoid applied)
    float*  X1   = (float*)(w + 112 * MBy);    // 32 MB
    __bf16* KCb  = (__bf16*)(w + 16 * MBy);    // 64 MB, reuses K/V after WKV
    float*  RCb  = (float*)(w + 80 * MBy);     // 32 MB, reuses R
    __bf16* WKb  = (__bf16*)(w + 144 * MBy);
    __bf16* WVb  = (__bf16*)(w + 146 * MBy);
    __bf16* WRb  = (__bf16*)(w + 148 * MBy);
    __bf16* WCKb = (__bf16*)(w + 150 * MBy);   // 8 MB
    __bf16* WCVb = (__bf16*)(w + 158 * MBy);   // 8 MB
    __bf16* WCRb = (__bf16*)(w + 166 * MBy);   // total 168 MB

    // --- weights to bf16 ---
    {
        const int n1 = CC * CC, n4 = 4 * CC * CC;
        dim3 b(256);
        cvt_bf16_kernel<<<dim3((n1 + 255) / 256), b, 0, stream>>>(Wk,  WKb,  n1);
        cvt_bf16_kernel<<<dim3((n1 + 255) / 256), b, 0, stream>>>(Wv,  WVb,  n1);
        cvt_bf16_kernel<<<dim3((n1 + 255) / 256), b, 0, stream>>>(Wr,  WRb,  n1);
        cvt_bf16_kernel<<<dim3((n4 + 255) / 256), b, 0, stream>>>(Wck, WCKb, n4);
        cvt_bf16_kernel<<<dim3((n4 + 255) / 256), b, 0, stream>>>(Wcv, WCVb, n4);
        cvt_bf16_kernel<<<dim3((n1 + 255) / 256), b, 0, stream>>>(Wcr, WCRb, n1);
    }

    // --- time mixing: LN1 + shift-mix -> bf16 activations ---
    ln_mix_kernel<<<dim3(MM), dim3(256), 0, stream>>>(x, ln1w, ln1b, ts, Abf, 0);

    // --- k, v, r = sigmoid(xxx @ Wr^T) ---
    dim3 g1(CC / TN, MM / TM);   // (8, 64)
    gemm_bf16_kernel<0><<<g1, dim3(256), 0, stream>>>(Abf, WKb, Kbuf, nullptr, nullptr, nullptr, CC, CC);
    gemm_bf16_kernel<0><<<g1, dim3(256), 0, stream>>>(Abf, WVb, Vbuf, nullptr, nullptr, nullptr, CC, CC);
    gemm_bf16_kernel<1><<<g1, dim3(256), 0, stream>>>(Abf, WRb, Rbuf, nullptr, nullptr, nullptr, CC, CC);

    // --- WKV scan + residual: X1 = x + sigmoid(r) * wkv ---
    wkv_kernel<<<dim3((BB * CC) / 256), dim3(256), 0, stream>>>(tf, Kbuf, Vbuf, Rbuf, x, X1);

    // --- channel mixing: LN2 + mix (opposite sign, t=0 wraps to T-1) ---
    ln_mix_kernel<<<dim3(MM), dim3(256), 0, stream>>>(X1, ln2w, ln2b, ts, Abf, 1);

    // --- kc = relu(xxx2 @ Wck^T)^2 (bf16), rc = sigmoid(xxx2 @ Wcr^T) ---
    dim3 g4(4 * CC / TN, MM / TM);  // (32, 64)
    gemm_bf16_kernel<2><<<g4, dim3(256), 0, stream>>>(Abf, WCKb, nullptr, KCb, nullptr, nullptr, 4 * CC, CC);
    gemm_bf16_kernel<1><<<g1, dim3(256), 0, stream>>>(Abf, WCRb, RCb, nullptr, nullptr, nullptr, CC, CC);

    // --- out = X1 + rc * (kc @ Wcv^T) ---
    gemm_bf16_kernel<3><<<g1, dim3(256), 0, stream>>>(KCb, WCVb, out, nullptr, X1, RCb, CC, 4 * CC);
}
